// TangentGNN_54176717472198
// MI455X (gfx1250) — matrix operand order repaired and checked by reading, not compiled, not run on hardware
//
#include <hip/hip_runtime.h>
#include <hip/hip_bf16.h>

typedef __bf16 v16bf __attribute__((ext_vector_type(16)));
typedef __bf16 v8bf  __attribute__((ext_vector_type(8)));
typedef __bf16 v4bf  __attribute__((ext_vector_type(4)));
typedef float  v8f   __attribute__((ext_vector_type(8)));

#define INDIM 128
#define HIDF  128
#define NCLS  64
#define KDIM  128   // K is 128 for all three GEMMs -> hardcode & fully unroll

// ---------------------------------------------------------------------------
// Pack weights W[K=128, Nout] (f32 row-major) into fragment-major bf16:
// P[((kb/32)*ntiles + nt)*512 + lane*16 + e] = bf16( W[kb + (lane>>4)*16 + e,
//                                                      nt*16 + (lane&15)] )
// so each lane's 16-element B fragment is one contiguous 32-byte load.
// ---------------------------------------------------------------------------
__global__ void k_pack_w(const float* __restrict__ W, __bf16* __restrict__ P, int Nout) {
    int i = blockIdx.x * blockDim.x + threadIdx.x;
    if (i >= KDIM * Nout) return;
    int e    = i & 15;
    int lane = (i >> 4) & 31;
    int frag = i >> 9;                 // 512 elements per fragment
    int ntiles = Nout >> 4;
    int kbc = frag / ntiles;
    int nt  = frag % ntiles;
    int k = kbc * 32 + (lane >> 4) * 16 + e;
    int n = nt * 16 + (lane & 15);
    P[i] = (__bf16)W[(size_t)k * Nout + n];
}

// f32 -> bf16 bulk convert (4 elements/thread)
__global__ void k_f32_to_bf16(const float* __restrict__ s, __bf16* __restrict__ d, int n4) {
    int i = blockIdx.x * blockDim.x + threadIdx.x;
    if (i >= n4) return;
    float4 v = ((const float4*)s)[i];
    v4bf o;
    o[0] = (__bf16)v.x; o[1] = (__bf16)v.y; o[2] = (__bf16)v.z; o[3] = (__bf16)v.w;
    ((v4bf*)d)[i] = o;
}

// ---------------------------------------------------------------------------
// Dense GEMM: C[M,Nout] = A[M,128] @ W (+ bias). A is bf16 row-major,
// W pre-packed fragment-major bf16, f32 accumulation via WMMA.
// blockDim = 128 (4 waves); wave w -> 16x16 tile (blockIdx.x*16,
// (blockIdx.y*4+w)*16). Grids divide exactly -> EXEC all-ones at every WMMA.
// ---------------------------------------------------------------------------
__global__ __launch_bounds__(128)
void gcn_gemm_wmma(const __bf16* __restrict__ A, const __bf16* __restrict__ P,
                   const float* __restrict__ bias, float* __restrict__ C, int Nout) {
    const int lane  = threadIdx.x & 31;
    const int wave  = threadIdx.x >> 5;
    const int lmod  = lane & 15;
    const int lhalf = lane >> 4;
    const int m0 = blockIdx.x * 16;
    const int nt = blockIdx.y * 4 + wave;          // N-tile index
    const int n0 = nt * 16;
    const int ntiles = Nout >> 4;

    const int row = m0 + lmod;
    const int col = n0 + lmod;

    v8f acc = {0.f, 0.f, 0.f, 0.f, 0.f, 0.f, 0.f, 0.f};

#pragma unroll
    for (int kbc = 0; kbc < KDIM / 32; ++kbc) {
        // A fragment: elems 0..15 -> K = kbc*32 + lhalf*16 + e
        const v8bf* ap = (const v8bf*)(A + (size_t)row * KDIM + kbc * 32 + lhalf * 16);
        v8bf alo = ap[0];
        v8bf ahi = ap[1];      // +8 bf16 elements
        v16bf afrag = __builtin_shufflevector(alo, ahi,
                        0,1,2,3,4,5,6,7,8,9,10,11,12,13,14,15);
        // B fragment: one contiguous 32B load from the packed layout
        v16bf bfrag = *(const v16bf*)(P + ((size_t)(kbc * ntiles + nt) * 32 + lane) * 16);

        acc = __builtin_amdgcn_wmma_f32_16x16x32_bf16(
            false, afrag, false, bfrag, (short)0, acc, false, false);
    }

    const float bval = bias ? bias[col] : 0.0f;
#pragma unroll
    for (int r = 0; r < 8; ++r)
        C[(size_t)(m0 + lhalf * 8 + r) * Nout + col] = acc[r] + bval;
}

// ---------------------------------------------------------------------------
// Degree / normalization helpers
// ---------------------------------------------------------------------------
__global__ void k_zero_u32(unsigned* __restrict__ p, int n) {
    int i = blockIdx.x * blockDim.x + threadIdx.x;
    if (i < n) p[i] = 0u;
}

__global__ void k_deg_count(const int* __restrict__ dst, unsigned* __restrict__ cnt, int E) {
    int i = blockIdx.x * blockDim.x + threadIdx.x;
    if (i < E) atomicAdd(&cnt[dst[i]], 1u);
}

__global__ void k_dinv(const unsigned* __restrict__ cnt, float* __restrict__ dinv, int n) {
    int i = blockIdx.x * blockDim.x + threadIdx.x;
    if (i < n) dinv[i] = rsqrtf((float)cnt[i] + 1.0f);   // +1 self-loop
}

// hacc[v,:] = tmp[v,:] * dinv[v]^2  (self-loop message; initializes accumulator)
__global__ void k_selfmsg(const float* __restrict__ t, const float* __restrict__ dinv,
                          float* __restrict__ h, int n) {
    int gid = blockIdx.x * blockDim.x + threadIdx.x;  // n*32 threads, 4 feats each
    int v = gid >> 5;
    if (v >= n) return;
    int c4 = (gid & 31) * 4;
    float w = dinv[v]; w *= w;
    float4 val = *(const float4*)(t + (size_t)v * HIDF + c4);
    float4 o;
    o.x = val.x * w; o.y = val.y * w; o.z = val.z * w; o.w = val.w * w;
    *(float4*)(h + (size_t)v * HIDF + c4) = o;
}

// One wave per edge: gather 128 floats of tmp[src], scale, scatter-add to hacc[dst].
__global__ __launch_bounds__(256)
void k_edge(const float* __restrict__ t, const int* __restrict__ src,
            const int* __restrict__ dst, const float* __restrict__ dinv,
            float* __restrict__ h, int E) {
    int gid = blockIdx.x * blockDim.x + threadIdx.x;
    int e = gid >> 5;
    if (e >= E) return;
    int lane = gid & 31;
    int s = src[e], d = dst[e];               // wave-uniform -> scalar loads
    float nrm = dinv[s] * dinv[d];
    float4 v = *(const float4*)(t + (size_t)s * HIDF + lane * 4);
    float* op = h + (size_t)d * HIDF + lane * 4;
    unsafeAtomicAdd(op + 0, v.x * nrm);
    unsafeAtomicAdd(op + 1, v.y * nrm);
    unsafeAtomicAdd(op + 2, v.z * nrm);
    unsafeAtomicAdd(op + 3, v.w * nrm);
}

// hbf[v,:] = bf16( relu(hacc[v,:] + b) )  — feeds the next GEMM directly.
__global__ void k_bias_relu_bf16(const float* __restrict__ h, const float* __restrict__ b,
                                 __bf16* __restrict__ o, int n) {
    int gid = blockIdx.x * blockDim.x + threadIdx.x;
    int v = gid >> 5;
    if (v >= n) return;
    int c4 = (gid & 31) * 4;
    float4 bb = *(const float4*)(b + c4);
    float4 x = *(const float4*)(h + (size_t)v * HIDF + c4);
    v4bf r;
    r[0] = (__bf16)fmaxf(x.x + bb.x, 0.f);
    r[1] = (__bf16)fmaxf(x.y + bb.y, 0.f);
    r[2] = (__bf16)fmaxf(x.z + bb.z, 0.f);
    r[3] = (__bf16)fmaxf(x.w + bb.w, 0.f);
    *(v4bf*)(o + (size_t)v * HIDF + c4) = r;
}

// ---------------------------------------------------------------------------
extern "C" void kernel_launch(void* const* d_in, const int* in_sizes, int n_in,
                              void* d_out, int out_size, void* d_ws, size_t ws_size,
                              hipStream_t stream) {
    const float* x  = (const float*)d_in[0];
    const int*   ei = (const int*)d_in[1];     // [2,E] int32
    const float* W1 = (const float*)d_in[2];
    const float* b1 = (const float*)d_in[3];
    const float* W2 = (const float*)d_in[4];
    const float* b2 = (const float*)d_in[5];
    const float* Wc = (const float*)d_in[6];
    const float* bc = (const float*)d_in[7];
    float* out = (float*)d_out;

    const int N = in_sizes[0] / INDIM;
    const int E = in_sizes[1] / 2;
    const int* src = ei;
    const int* dst = ei + E;

    // Workspace: tmp f32[N*128] | hacc f32[N*128] | dinv f32[N] | cnt u32[N]
    //            | wpack bf16[128*128] | hbf bf16[N*128]
    float*    tmp   = (float*)d_ws;
    float*    hacc  = tmp + (size_t)N * HIDF;
    float*    dinv  = hacc + (size_t)N * HIDF;
    unsigned* cnt   = (unsigned*)(dinv + N);
    __bf16*   wpack = (__bf16*)(cnt + N);
    __bf16*   hbf   = wpack + (size_t)KDIM * HIDF;

    const int TPB = 256;
    const int gN  = (N + TPB - 1) / TPB;
    const int gE  = (E + TPB - 1) / TPB;
    const int gNf = (N * 32 + TPB - 1) / TPB;   // node x float4-chunk threads
    const int gEw = (E * 32 + TPB - 1) / TPB;   // one wave per edge
    const int gCv = (N * HIDF / 4 + TPB - 1) / TPB;
    const int gPk = (KDIM * HIDF + TPB - 1) / TPB;

    dim3 gemmBlk(128, 1, 1);
    dim3 gridHid(N / 16, HIDF / 64, 1);
    dim3 gridCls(N / 16, NCLS / 64, 1);

    // --- degree / normalization ---
    k_zero_u32 <<<gN, TPB, 0, stream>>>(cnt, N);
    k_deg_count<<<gE, TPB, 0, stream>>>(dst, cnt, E);
    k_dinv     <<<gN, TPB, 0, stream>>>(cnt, dinv, N);

    // --- layer 1: hbf = bf16( relu( Agg(x @ W1) + b1 ) ) ---
    k_f32_to_bf16<<<gCv, TPB, 0, stream>>>(x, hbf, N * HIDF / 4);
    k_pack_w     <<<gPk, TPB, 0, stream>>>(W1, wpack, HIDF);
    gcn_gemm_wmma<<<gridHid, gemmBlk, 0, stream>>>(hbf, wpack, nullptr, tmp, HIDF);
    k_selfmsg    <<<gNf, TPB, 0, stream>>>(tmp, dinv, hacc, N);
    k_edge       <<<gEw, TPB, 0, stream>>>(tmp, src, dst, dinv, hacc, E);
    k_bias_relu_bf16<<<gNf, TPB, 0, stream>>>(hacc, b1, hbf, N);

    // --- layer 2 ---
    k_pack_w     <<<gPk, TPB, 0, stream>>>(W2, wpack, HIDF);
    gcn_gemm_wmma<<<gridHid, gemmBlk, 0, stream>>>(hbf, wpack, nullptr, tmp, HIDF);
    k_selfmsg    <<<gNf, TPB, 0, stream>>>(tmp, dinv, hacc, N);
    k_edge       <<<gEw, TPB, 0, stream>>>(tmp, src, dst, dinv, hacc, E);
    k_bias_relu_bf16<<<gNf, TPB, 0, stream>>>(hacc, b2, hbf, N);

    // --- classifier: out = hbf @ Wc + bc (f32 out) ---
    k_pack_w     <<<(KDIM * NCLS + TPB - 1) / TPB, TPB, 0, stream>>>(Wc, wpack, NCLS);
    gcn_gemm_wmma<<<gridCls, gemmBlk, 0, stream>>>(hbf, wpack, bc, out, NCLS);
}
